// PTQSLBatchingQuantMatMul_68839735820689
// MI455X (gfx1250) — matrix operand order repaired and checked by reading, not compiled, not run on hardware
//
#include <hip/hip_runtime.h>

typedef __attribute__((ext_vector_type(16))) _Float16 v16h;
typedef __attribute__((ext_vector_type(8)))  _Float16 v8h;
typedef __attribute__((ext_vector_type(4)))  _Float16 v4h;
typedef __attribute__((ext_vector_type(8)))  float    v8f;
typedef __attribute__((ext_vector_type(4)))  float    v4f;
typedef __attribute__((ext_vector_type(4)))  int      v4i;

#define SEQ  2048
#define WDIM 128
#define BM   128
#define BN   128
#define LDK  136   // padded row stride in halves: 272 B (16B-aligned, bank-spread)
#define NBH  32    // 2 batch * 16 heads

#if defined(__has_builtin)
#if __has_builtin(__builtin_amdgcn_global_load_async_to_lds_b128)
#define USE_ASYNC_LDS 1
#endif
#endif

__device__ __forceinline__ _Float16 fq1(float x, float s, float r) {
  float t = rintf(x * r);
  t = fminf(fmaxf(t, -128.0f), 127.0f);
  return (_Float16)(t * s);
}

#ifdef USE_ASYNC_LDS
typedef v4i __attribute__((address_space(1))) v4i_g;   // global AS pointee
typedef v4i __attribute__((address_space(3))) v4i_l;   // LDS AS pointee

__device__ __forceinline__ void cp16_async(void* l, const void* g) {
  // per-lane 16B global -> LDS via async DMA path (ASYNCcnt)
  __builtin_amdgcn_global_load_async_to_lds_b128(
      (v4i_g*)g, (v4i_l*)l, 0, 0);
}
#endif

// ---- shared WMMA compute + NT store on staged K-contiguous LDS tiles ----
__device__ __forceinline__ void wmma_compute_store(
    const _Float16 (*Ash)[LDK], const _Float16 (*Bsh)[LDK],
    float* __restrict__ Og, int m0, int n0) {
  const int tid  = threadIdx.x;
  const int lane = tid & 31;
  const int wave = tid >> 5;
  const int wm = (wave & 3) * 32;   // 4 wave-rows x 32 M
  const int wn = (wave >> 2) * 64;  // 2 wave-cols x 64 N

  v8f acc[2][4];
#pragma unroll
  for (int i = 0; i < 2; ++i)
#pragma unroll
    for (int j = 0; j < 4; ++j) acc[i][j] = (v8f){0,0,0,0,0,0,0,0};

  const int  hlane  = lane & 15;
  const bool hihalf = lane >= 16;

#pragma unroll
  for (int ks = 0; ks < 4; ++ks) {
    const int k0 = ks * 32;
    v16h afr[2], bfr[4];
    // A frag (16-bit 16x32 ISA layout): lanes 0-15: K {0..7,16..23}; 16-31: K {8..15,24..31}
#pragma unroll
    for (int i = 0; i < 2; ++i) {
      const _Float16* row = &Ash[wm + i * 16 + hlane][0];
      const int kb = k0 + (hihalf ? 8 : 0);
      v8h lo = *(const v8h*)(row + kb);
      v8h up = *(const v8h*)(row + kb + 16);
#pragma unroll
      for (int e = 0; e < 8; ++e) { afr[i][e] = lo[e]; afr[i][8 + e] = up[e]; }
    }
    // B frag (32x16): lane holds column n; lanes 0-15: K {0..7,16..23}; 16-31: K {8..15,24..31}
#pragma unroll
    for (int j = 0; j < 4; ++j) {
      const _Float16* row = &Bsh[wn + j * 16 + hlane][0];
      const int kb = k0 + (hihalf ? 8 : 0);
      v8h lo = *(const v8h*)(row + kb);
      v8h up = *(const v8h*)(row + kb + 16);
#pragma unroll
      for (int e = 0; e < 8; ++e) { bfr[j][e] = lo[e]; bfr[j][8 + e] = up[e]; }
    }
#pragma unroll
    for (int i = 0; i < 2; ++i)
#pragma unroll
      for (int j = 0; j < 4; ++j)
        acc[i][j] = __builtin_amdgcn_wmma_f32_16x16x32_f16(
            false, afr[i], false, bfr[j], (short)0, acc[i][j], false, false);
  }

  // C store per 16x16 f32 layout; non-temporal (537 MB stream-out, keep L2 for A/B)
  const int moff = hihalf ? 8 : 0;
#pragma unroll
  for (int i = 0; i < 2; ++i)
#pragma unroll
    for (int j = 0; j < 4; ++j) {
      float* op = Og + (size_t)(m0 + wm + i * 16 + moff) * SEQ
                     + (n0 + wn + j * 16 + hlane);
#pragma unroll
      for (int r = 0; r < 8; ++r)
        __builtin_nontemporal_store(acc[i][j][r], op + (size_t)r * SEQ);
    }
}

// ---------- phase 1a: fake-quant A -> f16, same layout ----------
__global__ __launch_bounds__(256, 1)
void fq_a_kernel(const float* __restrict__ A, const float* __restrict__ Aint,
                 _Float16* __restrict__ Aq) {
  const size_t e = ((size_t)blockIdx.x * 256 + threadIdx.x) * 4;  // float4 granule
  const int m  = (int)((e >> 7) & (SEQ - 1));
  const int bh = (int)(e >> 18);           // 2048*128 = 2^18 elems per (b,h)
  const int hg = (bh & 15) >> 2;
  const float s = Aint[hg * 4 + (m >> 9)]; // scale: (head group, 512-row group)
  const float r = 1.0f / s;
  v4f v = *(const v4f*)(A + e);
  v4h q;
#pragma unroll
  for (int j = 0; j < 4; ++j) q[j] = fq1(v[j], s, r);
  *(v4h*)(Aq + e) = q;
}

// ---------- phase 1b: fake-quant B -> f16, transposed Bt[bh][n][k] ----------
__global__ __launch_bounds__(256, 1)
void fq_bt_kernel(const float* __restrict__ B, const float* __restrict__ Bint,
                  _Float16* __restrict__ Bt) {
  __shared__ _Float16 T[64][72];  // 64x64 tile, +8 halves pad
  const int tid = threadIdx.x;
  const int bh  = blockIdx.z;
  const int hg  = (bh & 15) >> 2;
  const int k0  = blockIdx.y * 64;
  const int n0  = blockIdx.x * 64;
  const float* Bg = B + (size_t)bh * WDIM * SEQ;

  const int n4 = tid & 15, kk = tid >> 4;
#pragma unroll
  for (int p = 0; p < 4; ++p) {
    const int kr = p * 16 + kk;
    const int ka = k0 + kr;
    const float s = Bint[hg * 4 + (ka >> 5)];  // scale varies per 32-K group
    const float r = 1.0f / s;
    v4f v = *(const v4f*)(Bg + (size_t)ka * SEQ + n0 + n4 * 4);
#pragma unroll
    for (int j = 0; j < 4; ++j) T[n4 * 4 + j][kr] = fq1(v[j], s, r);
  }
  __syncthreads();

  _Float16* Bo = Bt + ((size_t)bh * SEQ + n0) * WDIM + k0;
  const int k4 = (tid & 15) * 4, nn = tid >> 4;
#pragma unroll
  for (int p = 0; p < 4; ++p) {
    const int n = p * 16 + nn;
    *(v4h*)(Bo + (size_t)n * WDIM + k4) = *(const v4h*)&T[n][k4];
  }
}

// ---------- phase 2: f16 GEMM, async global->LDS staging ----------
__global__ __launch_bounds__(256, 1)
void gemm_f16_kernel(const _Float16* __restrict__ Aq,
                     const _Float16* __restrict__ Bt,
                     float* __restrict__ Out) {
  __shared__ _Float16 Ash[BM][LDK];
  __shared__ _Float16 Bsh[BN][LDK];
  const int tid = threadIdx.x;
  const int bh  = blockIdx.z;
  const int m0  = blockIdx.y * BM;
  const int n0  = blockIdx.x * BN;

  const _Float16* Ag = Aq + ((size_t)bh * SEQ + m0) * WDIM;
  const _Float16* Bg = Bt + ((size_t)bh * SEQ + n0) * WDIM;

#pragma unroll
  for (int p = 0; p < 8; ++p) {
    const int c   = p * 256 + tid;      // 2048 chunks of 16B per tile
    const int row = c >> 4;
    const int col = (c & 15) * 8;
#ifdef USE_ASYNC_LDS
    cp16_async(&Ash[row][col], Ag + (size_t)row * WDIM + col);
    cp16_async(&Bsh[row][col], Bg + (size_t)row * WDIM + col);
#else
    *(v8h*)&Ash[row][col] = *(const v8h*)(Ag + (size_t)row * WDIM + col);
    *(v8h*)&Bsh[row][col] = *(const v8h*)(Bg + (size_t)row * WDIM + col);
#endif
  }
#ifdef USE_ASYNC_LDS
#if __has_builtin(__builtin_amdgcn_s_wait_asynccnt)
  __builtin_amdgcn_s_wait_asynccnt(0);
#else
  asm volatile("s_wait_asynccnt 0" ::: "memory");
#endif
#endif
  __syncthreads();

  wmma_compute_store(Ash, Bsh, Out + (size_t)bh * SEQ * SEQ, m0, n0);
}

// ---------- fallback: fully fused (used only if workspace too small) ----------
__global__ __launch_bounds__(256, 1)
void fused_kernel(const float* __restrict__ A, const float* __restrict__ B,
                  const float* __restrict__ Aint, const float* __restrict__ Bint,
                  float* __restrict__ Out) {
  __shared__ _Float16 Ash[BM][LDK];
  __shared__ _Float16 Bsh[BN][LDK];
  const int tid = threadIdx.x;
  const int bh = blockIdx.z;
  const int h  = bh & 15;
  const int hg = h >> 2;
  const int m0 = blockIdx.y * BM;
  const int n0 = blockIdx.x * BN;

  const float sA = Aint[hg * 4 + (m0 >> 9)];
  const float rA = 1.0f / sA;
  float sB[4], rB[4];
#pragma unroll
  for (int g = 0; g < 4; ++g) { sB[g] = Bint[hg * 4 + g]; rB[g] = 1.0f / sB[g]; }

  const float* Ag = A + ((size_t)bh * SEQ + m0) * WDIM;
  const float* Bg = B + (size_t)bh * WDIM * SEQ + n0;

  const int c4 = tid & 31, r0 = tid >> 5;
#pragma unroll
  for (int p = 0; p < 16; ++p) {
    const int m = p * 8 + r0;
    v4f v = *(const v4f*)(Ag + (size_t)m * WDIM + c4 * 4);
    v4h q;
#pragma unroll
    for (int j = 0; j < 4; ++j) q[j] = fq1(v[j], sA, rA);
    *(v4h*)&Ash[m][c4 * 4] = q;
  }
#pragma unroll
  for (int p = 0; p < 16; ++p) {
    const int k = p * 8 + r0;
    const int g = k >> 5;
    v4f v = *(const v4f*)(Bg + (size_t)k * SEQ + c4 * 4);
#pragma unroll
    for (int j = 0; j < 4; ++j) Bsh[c4 * 4 + j][k] = fq1(v[j], sB[g], rB[g]);
  }
  __syncthreads();

  wmma_compute_store(Ash, Bsh, Out + (size_t)bh * SEQ * SEQ, m0, n0);
}

extern "C" void kernel_launch(void* const* d_in, const int* in_sizes, int n_in,
                              void* d_out, int out_size, void* d_ws, size_t ws_size,
                              hipStream_t stream) {
  const float* A  = (const float*)d_in[0];
  const float* B  = (const float*)d_in[1];
  const float* Ai = (const float*)d_in[2];
  const float* Bi = (const float*)d_in[3];
  float* O = (float*)d_out;

  const size_t elems = (size_t)NBH * SEQ * WDIM;          // 8,388,608 per operand
  const size_t need  = 2 * elems * sizeof(_Float16);      // 33.5 MB

  if (ws_size >= need) {
    _Float16* Aq = (_Float16*)d_ws;
    _Float16* Bt = Aq + elems;
    fq_a_kernel<<<dim3((unsigned)(elems / 4 / 256)), dim3(256), 0, stream>>>(A, Ai, Aq);
    fq_bt_kernel<<<dim3(SEQ / 64, WDIM / 64, NBH), dim3(256), 0, stream>>>(B, Bi, Bt);
    gemm_f16_kernel<<<dim3(SEQ / BN, SEQ / BM, NBH), dim3(256), 0, stream>>>(Aq, Bt, O);
  } else {
    fused_kernel<<<dim3(SEQ / BN, SEQ / BM, NBH), dim3(256), 0, stream>>>(A, B, Ai, Bi, O);
  }
}
